// EDA_Attention_79731772883305
// MI455X (gfx1250) — matrix operand order repaired and checked by reading, not compiled, not run on hardware
//
#include <hip/hip_runtime.h>
#include <math.h>

// ---------------------------------------------------------------------------
// Problem constants (from reference): B=16, H=W=48, C=128, G=4, NH=8, SS=8,
// K=3, RED=4.  L = 2304, hd = 16, windows = 6 per batch (48x8 strips),
// tokens per window = 384.
// ---------------------------------------------------------------------------
#define BB   16
#define HH   48
#define WW   48
#define CC   128
#define GG   4
#define NHH  8
#define HD   16
#define LTOT (HH * WW)          // 2304
#define NWIN 6                  // W / SS
#define WTOK 384                // tokens per strip window
#define SCALE 0.25f             // hd^-0.5

#ifndef __has_builtin
#define __has_builtin(x) 0
#endif

typedef __bf16 bf16;
typedef bf16  v16bf __attribute__((ext_vector_type(16)));
typedef bf16  v8bfv __attribute__((ext_vector_type(8)));
typedef bf16  v4bfv __attribute__((ext_vector_type(4)));
typedef float v8f   __attribute__((ext_vector_type(8)));

// ---------------------------------------------------------------------------
// Cross-lane butterfly reductions over 16-lane rows (lanes 0-15 / 16-31
// reduce independently) using v_permlane16_b32 (pure VALU, no LDS traffic).
// Nibble j of the selector = source row-lane for row-lane j.
// ---------------------------------------------------------------------------
__device__ __forceinline__ float permrow16(float v, unsigned lo, unsigned hi) {
#if __has_builtin(__builtin_amdgcn_permlane16)
    int s = __builtin_bit_cast(int, v);
    s = __builtin_amdgcn_permlane16(s, s, (int)lo, (int)hi, false, false);
    return __builtin_bit_cast(float, s);
#else
    (void)lo; (void)hi;
    return v;
#endif
}

__device__ __forceinline__ float rowmax16(float x) {
#if __has_builtin(__builtin_amdgcn_permlane16)
    x = fmaxf(x, permrow16(x, 0x67452301u, 0xEFCDAB89u)); // xor 1
    x = fmaxf(x, permrow16(x, 0x54761032u, 0xDCFE98BAu)); // xor 2
    x = fmaxf(x, permrow16(x, 0x32107654u, 0xBA98FEDCu)); // xor 4
    x = fmaxf(x, permrow16(x, 0xFEDCBA98u, 0x76543210u)); // xor 8
#else
    for (int off = 1; off < 16; off <<= 1) x = fmaxf(x, __shfl_xor(x, off, 32));
#endif
    return x;
}

__device__ __forceinline__ float rowsum16(float x) {
#if __has_builtin(__builtin_amdgcn_permlane16)
    x += permrow16(x, 0x67452301u, 0xEFCDAB89u);          // xor 1
    x += permrow16(x, 0x54761032u, 0xDCFE98BAu);          // xor 2
    x += permrow16(x, 0x32107654u, 0xBA98FEDCu);          // xor 4
    x += permrow16(x, 0xFEDCBA98u, 0x76543210u);          // xor 8
#else
    for (int off = 1; off < 16; off <<= 1) x += __shfl_xor(x, off, 32);
#endif
    return x;
}

// ---------------------------------------------------------------------------
// Kernel A: adaptive avg pool 48x48 -> 3x3 (uniform 16x16 block means).
// grid (9 regions, 16 batches), block 128 (one thread per channel).
// pooled[(b*128+c)*9 + r]
// ---------------------------------------------------------------------------
__global__ void pool_kernel(const float* __restrict__ v, float* __restrict__ pooled) {
    const int r  = blockIdx.x;      // 0..8
    const int b  = blockIdx.y;      // 0..15
    const int c  = threadIdx.x;     // 0..127
    const int ky = r / 3, kx = r % 3;
    float s = 0.f;
#pragma unroll 4
    for (int i = 0; i < 256; ++i) {
        const int h = ky * 16 + (i >> 4);
        const int w = kx * 16 + (i & 15);
        s += v[(b * LTOT + h * WW + w) * CC + c];
    }
    pooled[(b * CC + c) * 9 + r] = s * (1.f / 256.f);
}

// ---------------------------------------------------------------------------
// Kernel B: proj(1x1 conv -> BN eval -> exact GELU -> 1x1 conv) + softmax over
// G, producing dynamic depthwise weights (9 pooled positions) and dynamic bias
// (global-mean position).  grid = B*10, block = 128.
// ---------------------------------------------------------------------------
__global__ void dynparam_kernel(const float* __restrict__ pooled,
                                const float* __restrict__ p1w, const float* __restrict__ p1b,
                                const float* __restrict__ gam, const float* __restrict__ bet,
                                const float* __restrict__ rm,  const float* __restrict__ rv,
                                const float* __restrict__ p2w, const float* __restrict__ p2b,
                                const float* __restrict__ dynw, const float* __restrict__ dynb,
                                float* __restrict__ wdyn, float* __restrict__ biasb) {
    const int b = blockIdx.x / 10;
    const int p = blockIdx.x % 10;      // 0..8 pooled pos, 9 = global mean
    const int c = threadIdx.x;          // 0..127
    __shared__ float xsh[CC];
    __shared__ float hsh[32];

    float xv;
    if (p < 9) {
        xv = pooled[(b * CC + c) * 9 + p];
    } else {
        float s = 0.f;
#pragma unroll
        for (int r2 = 0; r2 < 9; ++r2) s += pooled[(b * CC + c) * 9 + r2];
        xv = s * (1.f / 9.f);
    }
    xsh[c] = xv;
    __syncthreads();

    if (c < 32) {
        float acc = p1b[c];
        for (int j = 0; j < CC; ++j) acc += p1w[c * CC + j] * xsh[j];
        const float inv = rsqrtf(rv[c] + 1e-5f);
        acc = (acc - rm[c]) * (inv * gam[c]) + bet[c];
        acc = 0.5f * acc * (1.f + erff(acc * 0.70710678118654752440f)); // exact GELU
        hsh[c] = acc;
    }
    __syncthreads();

    float y[GG];
#pragma unroll
    for (int gi = 0; gi < GG; ++gi) {
        const int o = gi * CC + c;
        float acc = p2b[o];
        for (int j = 0; j < 32; ++j) acc += p2w[o * 32 + j] * hsh[j];
        y[gi] = acc;
    }
    float mx = fmaxf(fmaxf(y[0], y[1]), fmaxf(y[2], y[3]));
    float ex[GG]; float es = 0.f;
#pragma unroll
    for (int gi = 0; gi < GG; ++gi) { ex[gi] = expf(y[gi] - mx); es += ex[gi]; }
    const float invs = 1.f / es;

    if (p < 9) {
        float acc = 0.f;
#pragma unroll
        for (int gi = 0; gi < GG; ++gi)
            acc += ex[gi] * invs * dynw[(gi * CC + c) * 9 + p];
        wdyn[(b * CC + c) * 9 + p] = acc;
    } else {
        float acc = 0.f;
#pragma unroll
        for (int gi = 0; gi < GG; ++gi)
            acc += ex[gi] * invs * dynb[gi * CC + c];
        biasb[b * CC + c] = acc;
    }
}

// ---------------------------------------------------------------------------
// Kernel C: per-(b,c) dynamic depthwise 3x3 conv over v_img + dynamic bias.
// Writes the positional term DIRECTLY into d_out (layouts coincide:
// out[b, h*48+w, c] == pos[b, c, h, w]).  Fully overwrites d_out each call.
// grid = B*H*W*C/256, block 256 (c fastest -> coalesced).
// ---------------------------------------------------------------------------
__global__ void dwconv_kernel(const float* __restrict__ v,
                              const float* __restrict__ wdyn,
                              const float* __restrict__ biasb,
                              float* __restrict__ out) {
    const int idx = blockIdx.x * 256 + threadIdx.x;
    const int c = idx & (CC - 1);
    int rest = idx >> 7;
    const int w = rest % WW; rest /= WW;
    const int h = rest % HH;
    const int b = rest / HH;            // uniform within a block

    __shared__ float wsh[CC * 9];
    __shared__ float bsh[CC];
    for (int i = threadIdx.x; i < CC * 9; i += 256) wsh[i] = wdyn[b * CC * 9 + i];
    if (threadIdx.x < CC) bsh[threadIdx.x] = biasb[b * CC + threadIdx.x];
    __syncthreads();

    float acc = bsh[c];
#pragma unroll
    for (int dy = 0; dy < 3; ++dy) {
#pragma unroll
        for (int dx = 0; dx < 3; ++dx) {
            const int hh2 = h + dy - 1;
            const int ww2 = w + dx - 1;
            if (hh2 >= 0 && hh2 < HH && ww2 >= 0 && ww2 < WW)
                acc += v[(b * LTOT + hh2 * WW + ww2) * CC + c] * wsh[c * 9 + dy * 3 + dx];
        }
    }
    out[idx] = acc;   // pos term; attention kernel accumulates on top
}

// ---------------------------------------------------------------------------
// Kernel D: strip attention with WMMA (flash-attention online softmax).
// grid (96 windows, 8 heads), block 256 = 8 waves.  Each wave owns 3 Q tiles
// of 16 rows; KV processed in 12 blocks of 32 tokens.
//   S  : v_wmma_f32_16x16x32_bf16, A = Q*scale (K padded 16->32), B = K^T
//        (A elems 8..15 are zero, so B rows 16..31 need no masking)
//   O += v_wmma_f32_16x16x32_bf16, A = P (full K=32 tokens),  B = V block
// P converted C-layout -> A-layout through a per-wave LDS scratch.
// Softmax row reductions via v_permlane16_b32 butterflies (no LDS).
// ---------------------------------------------------------------------------
__global__ __launch_bounds__(256)
void attn_kernel(const float* __restrict__ q, const float* __restrict__ k,
                 const float* __restrict__ v, float* __restrict__ out) {
    const int win  = blockIdx.x;        // 0..95 : b*6 + wx
    const int head = blockIdx.y;        // 0..7
    const int b    = win / NWIN;
    const int wx   = win % NWIN;
    const int tid  = threadIdx.x;
    const int wave = tid >> 5;
    const int lane = tid & 31;

    __shared__ bf16 Kl[WTOK * HD];      // K row-major [token][d]   12 KB
    __shared__ bf16 Vt[HD * WTOK];      // V transposed [d][token]  12 KB
    __shared__ bf16 Pb[8 * 16 * 32];    // per-wave P scratch        8 KB

    const int base = (b * LTOT) * CC + head * HD;

    // ---- stage K (row-major) and V (transposed) into LDS as bf16 ----
    // float4-vectorized: each step covers 4 consecutive dims of one token.
    for (int i = tid; i < (WTOK * HD) / 4; i += 256) {
        const int t = i >> 2;
        const int d = (i & 3) * 4;
        const int l = (t >> 3) * WW + wx * 8 + (t & 7);   // image linear index
        const float4 kk = *(const float4*)(k + base + l * CC + d);
        const float4 vv = *(const float4*)(v + base + l * CC + d);
        v4bfv kp; kp[0] = (bf16)kk.x; kp[1] = (bf16)kk.y;
                  kp[2] = (bf16)kk.z; kp[3] = (bf16)kk.w;
        *(v4bfv*)(Kl + t * HD + d) = kp;                  // ds_store_b64
        Vt[(d + 0) * WTOK + t] = (bf16)vv.x;
        Vt[(d + 1) * WTOK + t] = (bf16)vv.y;
        Vt[(d + 2) * WTOK + t] = (bf16)vv.z;
        Vt[(d + 3) * WTOK + t] = (bf16)vv.w;
    }
    __syncthreads();

    const int row = lane & 15;          // A-layout row / C-layout column
    const int hf  = lane >> 4;          // lane half (K-split in A/B layouts)
    bf16* Pw = Pb + wave * 512;

    for (int qt = wave; qt < 24; qt += 8) {
        // ---- load Q tile into WMMA A layout (bf16, pre-scaled, K pad->32) --
        const int tq = qt * 16 + row;
        const int lq = (tq >> 3) * WW + wx * 8 + (tq & 7);
        const float4* qp = (const float4*)(q + base + lq * CC + hf * 8);
        const float4 qa = qp[0], qb = qp[1];
        if (qt + 8 < 24) {              // prefetch next Q tile row
            const int tqn = (qt + 8) * 16 + row;
            const int lqn = (tqn >> 3) * WW + wx * 8 + (tqn & 7);
            __builtin_prefetch(q + base + lqn * CC + hf * 8, 0, 3);
        }
        v16bf a;
        a[0] = (bf16)(qa.x * SCALE); a[1] = (bf16)(qa.y * SCALE);
        a[2] = (bf16)(qa.z * SCALE); a[3] = (bf16)(qa.w * SCALE);
        a[4] = (bf16)(qb.x * SCALE); a[5] = (bf16)(qb.y * SCALE);
        a[6] = (bf16)(qb.z * SCALE); a[7] = (bf16)(qb.w * SCALE);
#pragma unroll
        for (int e = 8; e < 16; ++e) a[e] = (bf16)0.f;

        v8f O; float mrun[8], lrun[8];
#pragma unroll
        for (int r2 = 0; r2 < 8; ++r2) { O[r2] = 0.f; mrun[r2] = -3.0e38f; lrun[r2] = 0.f; }

#pragma unroll 2
        for (int blk = 0; blk < 12; ++blk) {
            const int t0 = blk * 32;

            // ---- B operands: K^T tokens [t0,t0+16) / [t0+16,t0+32).
            // B elems 8..15 (K rows 16..31) of every lane multiply A's zero
            // padding, so no masking needed -- all lanes load freely.
            v16bf b0, b1;
            {
                const v8bfv* k0 = (const v8bfv*)(Kl + (t0 + row) * HD);
                const v8bfv* k1 = (const v8bfv*)(Kl + (t0 + 16 + row) * HD);
                const v8bfv x0 = k0[0], x1 = k0[1], y0 = k1[0], y1 = k1[1];
#pragma unroll
                for (int e = 0; e < 8; ++e) {
                    b0[e] = hf ? x1[e] : x0[e]; b0[8 + e] = hf ? x0[e] : x1[e];
                    b1[e] = hf ? y1[e] : y0[e]; b1[8 + e] = hf ? y0[e] : y1[e];
                }
            }
            v8f z;
#pragma unroll
            for (int r2 = 0; r2 < 8; ++r2) z[r2] = 0.f;

            v8f s0 = __builtin_amdgcn_wmma_f32_16x16x32_bf16(false, a, false, b0,
                                                             (short)0, z, false, false);
            v8f s1 = __builtin_amdgcn_wmma_f32_16x16x32_bf16(false, a, false, b1,
                                                             (short)0, z, false, false);

            // ---- online softmax over this 32-token block (permlane VALU) ---
            float bm[8];
#pragma unroll
            for (int r2 = 0; r2 < 8; ++r2) bm[r2] = fmaxf(s0[r2], s1[r2]);
#pragma unroll
            for (int r2 = 0; r2 < 8; ++r2) bm[r2] = rowmax16(bm[r2]);

            float rs[8];
#pragma unroll
            for (int r2 = 0; r2 < 8; ++r2) {
                const float mn = fmaxf(mrun[r2], bm[r2]);
                const float al = __expf(mrun[r2] - mn);
                s0[r2] = __expf(s0[r2] - mn);
                s1[r2] = __expf(s1[r2] - mn);
                rs[r2] = s0[r2] + s1[r2];
                O[r2] *= al;
                lrun[r2] *= al;
                mrun[r2] = mn;
            }
#pragma unroll
            for (int r2 = 0; r2 < 8; ++r2) rs[r2] = rowsum16(rs[r2]);
#pragma unroll
            for (int r2 = 0; r2 < 8; ++r2) lrun[r2] += rs[r2];

            // ---- C-layout -> A-layout for P via per-wave LDS scratch ----
#pragma unroll
            for (int r2 = 0; r2 < 8; ++r2) {
                const int mr = r2 + hf * 8;
                Pw[mr * 32 + row]      = (bf16)s0[r2];
                Pw[mr * 32 + 16 + row] = (bf16)s1[r2];
            }
            asm volatile("s_wait_dscnt 0" ::: "memory");  // wave-local RAW on LDS

            v16bf ap;
            {
                const v8bfv p0 = *(const v8bfv*)(Pw + row * 32 + hf * 8);
                const v8bfv p1 = *(const v8bfv*)(Pw + row * 32 + 16 + hf * 8);
#pragma unroll
                for (int e = 0; e < 8; ++e) { ap[e] = p0[e]; ap[8 + e] = p1[e]; }
            }
            // ---- B operand: V block (32 tokens x 16 dims), full K used ----
            v16bf bv;
            {
                const bf16* vp = Vt + row * WTOK + t0 + hf * 8;
                const v8bfv v0 = *(const v8bfv*)vp, v1 = *(const v8bfv*)(vp + 16);
#pragma unroll
                for (int e = 0; e < 8; ++e) { bv[e] = v0[e]; bv[8 + e] = v1[e]; }
            }
            O = __builtin_amdgcn_wmma_f32_16x16x32_bf16(false, ap, false, bv,
                                                        (short)0, O, false, false);
        }

        // ---- epilogue: normalize, add positional term already in d_out ----
#pragma unroll
        for (int r2 = 0; r2 < 8; ++r2) {
            const int mr  = r2 + hf * 8;
            const int tq2 = qt * 16 + mr;
            const int lq2 = (tq2 >> 3) * WW + wx * 8 + (tq2 & 7);
            const int oi  = base + lq2 * CC + row;
            const float val = O[r2] / lrun[r2];
            out[oi] = out[oi] + val;
        }
    }
}

// ---------------------------------------------------------------------------
// Host launcher
// ---------------------------------------------------------------------------
extern "C" void kernel_launch(void* const* d_in, const int* in_sizes, int n_in,
                              void* d_out, int out_size, void* d_ws, size_t ws_size,
                              hipStream_t stream) {
    (void)in_sizes; (void)n_in; (void)out_size; (void)ws_size;
    const float* q    = (const float*)d_in[0];
    const float* k    = (const float*)d_in[1];
    const float* v    = (const float*)d_in[2];
    const float* p1w  = (const float*)d_in[3];
    const float* p1b  = (const float*)d_in[4];
    const float* gam  = (const float*)d_in[5];
    const float* bet  = (const float*)d_in[6];
    const float* rm   = (const float*)d_in[7];
    const float* rv   = (const float*)d_in[8];
    const float* p2w  = (const float*)d_in[9];
    const float* p2b  = (const float*)d_in[10];
    const float* dynw = (const float*)d_in[11];
    const float* dynb = (const float*)d_in[12];
    float* out = (float*)d_out;

    float* ws     = (float*)d_ws;
    float* pooled = ws;                       // B*C*9 = 18432 floats
    float* wdyn   = ws + BB * CC * 9;         // B*C*9 = 18432 floats
    float* biasb  = ws + 2 * BB * CC * 9;     // B*C   =  2048 floats

    pool_kernel<<<dim3(9, BB), 128, 0, stream>>>(v, pooled);
    dynparam_kernel<<<BB * 10, 128, 0, stream>>>(pooled, p1w, p1b, gam, bet, rm, rv,
                                                 p2w, p2b, dynw, dynb, wdyn, biasb);
    dwconv_kernel<<<(BB * LTOT * CC) / 256, 256, 0, stream>>>(v, wdyn, biasb, out);
    attn_kernel<<<dim3(BB * NWIN, NHH), 256, 0, stream>>>(q, k, v, out);
}
